// Attention_31559419691443
// MI455X (gfx1250) — matrix operand order repaired and checked by reading, not compiled, not run on hardware
//
#include <hip/hip_runtime.h>
#include <hip/hip_bf16.h>

// ---------------------------------------------------------------------------
// Attention (B=4, C=256, L=2048, H=8, D=64) for gfx1250 (MI455X, wave32).
// ~43 GFLOP, ~40MB traffic -> compute-bound: bf16 WMMA, f32 accumulate.
// Flash-attention kernel uses GLOBAL_LOAD_ASYNC_TO_LDS_B128 (ASYNCcnt) with
// double-buffered LDS K/V tiles: DMA of tile i+1 overlaps WMMA on tile i.
// ---------------------------------------------------------------------------

#define BATCH   4
#define CIN     256
#define SEQ     2048
#define HEADS   8
#define DHEAD   64
#define HIDDEN  512
#define QKV_O   1536
#define SCALE_Q 0.125f

typedef __attribute__((ext_vector_type(16))) __bf16 v16bf;
typedef __attribute__((ext_vector_type(8)))  __bf16 v8bf;
typedef __attribute__((ext_vector_type(8)))  float  v8f;

__device__ __forceinline__ v8f wmma_bf16(v16bf a, v16bf b, v8f c) {
  return __builtin_amdgcn_wmma_f32_16x16x32_bf16(false, a, false, b, (short)0, c,
                                                 false, false);
}

__device__ __forceinline__ v16bf cat8(v8bf lo, v8bf hi) {
  return __builtin_shufflevector(lo, hi, 0, 1, 2, 3, 4, 5, 6, 7,
                                 8, 9, 10, 11, 12, 13, 14, 15);
}
// B-operand: groups p..p+7 (K=kb..kb+7) and p+16..p+23 (K=kb+16..kb+23)
__device__ __forceinline__ v16bf ldB16(const __bf16* p) {
  return cat8(*(const v8bf*)p, *(const v8bf*)(p + 16));
}
// A-operand: groups p..p+7 (K=kb..kb+7) and p+16..p+23 (K=kb+16..kb+23)
__device__ __forceinline__ v16bf ldA16(const __bf16* p) {
  return cat8(*(const v8bf*)p, *(const v8bf*)(p + 16));
}

// 32B memory -> LDS, no VGPR round-trip, tracked by ASYNCcnt.
// LDS address = low 32 bits of the generic shared pointer (ISA 10.2).
__device__ __forceinline__ void async_cp32(const __bf16* gsrc, __bf16* ldst) {
  unsigned lds = (unsigned)(unsigned long long)ldst;
  asm volatile(
      "global_load_async_to_lds_b128 %0, %1, off\n\t"
      "global_load_async_to_lds_b128 %0, %1, off offset:16"
      :: "v"(lds), "v"(gsrc) : "memory");
}
__device__ __forceinline__ void wait_async0(void) {
  asm volatile("s_wait_asynccnt 0x0" ::: "memory");
}
__device__ __forceinline__ void wave_lds_fence(void) {  // wave-local DS RAW
  asm volatile("s_wait_dscnt 0x0" ::: "memory");
}

// Lane roles (ISA 7.12.2, wave32):
//   A: m = lane&15, kbA = (lane<16)?0:8
//   B: n = lane&15, kbB = (lane<16)?0:8
//   C/D: col n = lane&15, row m = r + ((lane<16)?0:8) for VGPR r

// ---------------------------------------------------------------------------
// Kernel 0a: f32 -> bf16 elementwise (weights)
// ---------------------------------------------------------------------------
__global__ void cvt_bf16_kernel(const float* __restrict__ s,
                                __bf16* __restrict__ d, int n) {
  int i = blockIdx.x * blockDim.x + threadIdx.x;
  if (i < n) d[i] = (__bf16)s[i];
}

// ---------------------------------------------------------------------------
// Kernel 0b: xT[b][l][c] = bf16(x[b][c][l])  (LDS-tiled 32x32 transpose)
// ---------------------------------------------------------------------------
__global__ void x_transpose_kernel(const float* __restrict__ x,
                                   __bf16* __restrict__ xT) {
  __shared__ float t[32][33];
  const int b  = blockIdx.z;
  const int c0 = blockIdx.y * 32;
  const int l0 = blockIdx.x * 32;
  const int tx = threadIdx.x, ty = threadIdx.y;  // 32 x 8
#pragma unroll
  for (int i = 0; i < 4; ++i)
    t[ty + 8 * i][tx] = x[((size_t)b * CIN + c0 + ty + 8 * i) * SEQ + l0 + tx];
  __syncthreads();
#pragma unroll
  for (int i = 0; i < 4; ++i)
    xT[((size_t)b * SEQ + l0 + ty + 8 * i) * CIN + c0 + tx] =
        (__bf16)t[tx][ty + 8 * i];
}

// ---------------------------------------------------------------------------
// Kernel 1: qkv projection. Per wave: 16 (o) x 64 (l) output tile.
// Q -> [b][h][l][d] (scaled), K -> [b][h][l][d], V -> [b][h][d][l] (transposed).
// ---------------------------------------------------------------------------
__global__ void qkv_proj_kernel(const __bf16* __restrict__ xT,
                                const __bf16* __restrict__ wq,
                                __bf16* __restrict__ Qw,
                                __bf16* __restrict__ Kw,
                                __bf16* __restrict__ Vt) {
  const int lane = threadIdx.x & 31;
  const int wave = threadIdx.x >> 5;
  int tile = blockIdx.x * 4 + wave;          // 12288 wave-tiles
  const int lg = tile & 31;  tile >>= 5;
  const int ot = tile % 96;  const int b = tile / 96;
  const int o0 = ot * 16, l0 = lg * 64;

  const int m   = lane & 15;
  const int kbA = (lane < 16) ? 0 : 8;
  const int nn  = lane & 15;
  const int kbB = (lane < 16) ? 0 : 8;

  v8f acc[4] = {{}, {}, {}, {}};
#pragma unroll
  for (int c0 = 0; c0 < CIN; c0 += 32) {
    const v16bf a = ldA16(wq + (size_t)(o0 + m) * CIN + c0 + kbA);
#pragma unroll
    for (int j = 0; j < 4; ++j) {
      const v16bf bm =
          ldB16(xT + ((size_t)b * SEQ + l0 + j * 16 + nn) * CIN + c0 + kbB);
      acc[j] = wmma_bf16(a, bm, acc[j]);
    }
  }

  const int mb   = (lane < 16) ? 0 : 8;
  const int part = o0 >> 9;               // 0=Q 1=K 2=V (uniform per tile)
  const int h    = (o0 >> 6) & 7;
  const int d0   = (o0 & 63) + mb;        // multiple of 8 -> 16B-aligned stores
  if (part < 2) {
    __bf16* dst = (part == 0) ? Qw : Kw;
    const float scl = (part == 0) ? SCALE_Q : 1.0f;
#pragma unroll
    for (int j = 0; j < 4; ++j) {
      const int l = l0 + j * 16 + nn;
      v8bf pk;
#pragma unroll
      for (int r = 0; r < 8; ++r) pk[r] = (__bf16)(acc[j][r] * scl);
      *(v8bf*)(dst + ((size_t)(b * HEADS + h) * SEQ + l) * DHEAD + d0) = pk;
    }
  } else {
#pragma unroll
    for (int j = 0; j < 4; ++j) {
      const int l = l0 + j * 16 + nn;
#pragma unroll
      for (int r = 0; r < 8; ++r)
        Vt[((size_t)(b * HEADS + h) * DHEAD + d0 + r) * SEQ + l] =
            (__bf16)acc[j][r];
    }
  }
}

// ---------------------------------------------------------------------------
// Kernel 2: flash attention. 4 waves/block share one (b,h); each wave owns a
// 16-query tile. K/V 32-key tiles double-buffered in LDS via async DMA:
// tile i+1 streams in (ASYNCcnt) while WMMAs consume tile i. One barrier/step.
// ---------------------------------------------------------------------------
__global__ void flash_attn_kernel(const __bf16* __restrict__ Qw,
                                  const __bf16* __restrict__ Kw,
                                  const __bf16* __restrict__ Vt,
                                  __bf16* __restrict__ attT) {
  const int lane = threadIdx.x & 31;
  const int wave = threadIdx.x >> 5;
  const int t    = threadIdx.x;
  int id = blockIdx.x;                   // 1024 blocks
  const int g = id & 31;  id >>= 5;
  const int h = id & 7;   const int b = id >> 3;
  const int q0 = (g * 4 + wave) * 16;

  const __bf16* Qbh = Qw + (size_t)(b * HEADS + h) * SEQ * DHEAD;
  const __bf16* Kbh = Kw + (size_t)(b * HEADS + h) * SEQ * DHEAD;
  const __bf16* Vbh = Vt + (size_t)(b * HEADS + h) * DHEAD * SEQ;  // [d][l]

  const int m   = lane & 15;
  const int kbA = (lane < 16) ? 0 : 8;
  const int nn  = lane & 15;
  const int kbB = (lane < 16) ? 0 : 8;
  const int mb  = (lane < 16) ? 0 : 8;

  // Double-buffered, padded LDS tiles (16B-aligned rows, conflict-free).
  __shared__ __bf16 Kt[2][32][72];       // [buf][key][dim]
  __shared__ __bf16 Vs[2][64][40];       // [buf][dim][key]
  __shared__ __bf16 Pl[4][16][32];       // per-wave P staging

  // cooperative async-staging roles (128 threads, 32B each per tile)
  const int krow = t >> 2, kseg = (t & 3) * 16;  // K: 32 rows x 128B
  const int vrow = t >> 1, vseg = (t & 1) * 16;  // V: 64 rows x 64B

  // prologue: start DMA of tile 0 into buffer 0
  async_cp32(Kbh + (size_t)krow * DHEAD + kseg, &Kt[0][krow][kseg]);
  async_cp32(Vbh + (size_t)vrow * SEQ + vseg, &Vs[0][vrow][vseg]);

  // Query tile (16x64): two A operands, loaded once (overlaps with DMA).
  const __bf16* qp = Qbh + (size_t)(q0 + m) * DHEAD + kbA;
  const v16bf aQ0 = ldA16(qp);
  const v16bf aQ1 = ldA16(qp + 32);

  v8f o_acc[4] = {{}, {}, {}, {}};
  float run_max[8], run_sum[8];
#pragma unroll
  for (int r = 0; r < 8; ++r) { run_max[r] = -3.0e38f; run_sum[r] = 0.0f; }

  for (int i = 0; i < SEQ / 32; ++i) {
    const int cur = i & 1, nxt = cur ^ 1;
    wait_async0();        // this wave's DMA of tile i has landed in LDS
    __syncthreads();      // all waves' DMA landed; prior reads of buf nxt done

    if (i + 1 < SEQ / 32) {  // kick DMA of tile i+1 into the other buffer
      const int k1 = (i + 1) * 32;
      async_cp32(Kbh + (size_t)(k1 + krow) * DHEAD + kseg, &Kt[nxt][krow][kseg]);
      async_cp32(Vbh + (size_t)vrow * SEQ + k1 + vseg, &Vs[nxt][vrow][vseg]);
    }

    // ---- S = Q K^T (16 x 32): two f32 accumulators ----------------------
    v8f s0 = {}, s1 = {};
    s0 = wmma_bf16(aQ0, ldB16(&Kt[cur][nn][kbB]), s0);
    s0 = wmma_bf16(aQ1, ldB16(&Kt[cur][nn][32 + kbB]), s0);
    s1 = wmma_bf16(aQ0, ldB16(&Kt[cur][nn + 16][kbB]), s1);
    s1 = wmma_bf16(aQ1, ldB16(&Kt[cur][nn + 16][32 + kbB]), s1);

    // ---- online softmax (row reductions inside 16-lane half-groups) -----
#pragma unroll
    for (int r = 0; r < 8; ++r) {
      float tmax = fmaxf(s0[r], s1[r]);
#pragma unroll
      for (int msk = 1; msk < 16; msk <<= 1)
        tmax = fmaxf(tmax, __shfl_xor(tmax, msk, 32));
      const float nmax = fmaxf(run_max[r], tmax);
      const float f    = __expf(run_max[r] - nmax);
      const float p0   = __expf(s0[r] - nmax);
      const float p1   = __expf(s1[r] - nmax);
      float tsum = p0 + p1;
#pragma unroll
      for (int msk = 1; msk < 16; msk <<= 1)
        tsum += __shfl_xor(tsum, msk, 32);
      run_sum[r] = run_sum[r] * f + tsum;
      run_max[r] = nmax;
#pragma unroll
      for (int j = 0; j < 4; ++j) o_acc[j][r] *= f;
      Pl[wave][mb + r][nn]      = (__bf16)p0;
      Pl[wave][mb + r][nn + 16] = (__bf16)p1;
    }
    // Pl is wave-private: wave-local in-order DS + dscnt wait is enough.
    wave_lds_fence();

    // ---- O += P(16x32) * V(32x64) ---------------------------------------
    const v16bf aP = ldA16(&Pl[wave][m][kbA]);
#pragma unroll
    for (int j = 0; j < 4; ++j)
      o_acc[j] = wmma_bf16(aP, ldB16(&Vs[cur][j * 16 + nn][kbB]), o_acc[j]);
  }

  // ---- normalize, scatter to attT[b][l][h*64+d] ---------------------------
  float inv[8];
#pragma unroll
  for (int r = 0; r < 8; ++r) inv[r] = 1.0f / run_sum[r];
#pragma unroll
  for (int r = 0; r < 8; ++r) {
    const size_t row = ((size_t)b * SEQ + q0 + mb + r) * HIDDEN + h * DHEAD;
#pragma unroll
    for (int j = 0; j < 4; ++j)
      attT[row + j * 16 + nn] = (__bf16)(o_acc[j][r] * inv[r]);
  }
}

// ---------------------------------------------------------------------------
// Kernel 3: y[b] = w_out(256x512) * att + bias. Per wave: 16(o) x 64(l).
// ---------------------------------------------------------------------------
__global__ void out_proj_kernel(const __bf16* __restrict__ attT,
                                const __bf16* __restrict__ wo,
                                const float* __restrict__ b_out,
                                float* __restrict__ y) {
  const int lane = threadIdx.x & 31;
  const int wave = threadIdx.x >> 5;
  int tile = blockIdx.x * 4 + wave;          // 2048 wave-tiles
  const int lg = tile & 31;  tile >>= 5;
  const int ot = tile & 15;  const int b = tile >> 4;
  const int o0 = ot * 16, l0 = lg * 64;

  const int m   = lane & 15;
  const int kbA = (lane < 16) ? 0 : 8;
  const int nn  = lane & 15;
  const int kbB = (lane < 16) ? 0 : 8;

  v8f acc[4] = {{}, {}, {}, {}};
#pragma unroll
  for (int c0 = 0; c0 < HIDDEN; c0 += 32) {
    const v16bf a = ldA16(wo + (size_t)(o0 + m) * HIDDEN + c0 + kbA);
#pragma unroll
    for (int j = 0; j < 4; ++j) {
      const v16bf bm =
          ldB16(attT + ((size_t)b * SEQ + l0 + j * 16 + nn) * HIDDEN + c0 + kbB);
      acc[j] = wmma_bf16(a, bm, acc[j]);
    }
  }

  const int mb = (lane < 16) ? 0 : 8;
  float bias[8];
#pragma unroll
  for (int r = 0; r < 8; ++r) bias[r] = b_out[o0 + mb + r];
#pragma unroll
  for (int j = 0; j < 4; ++j) {
    const int l = l0 + j * 16 + nn;
#pragma unroll
    for (int r = 0; r < 8; ++r)
      y[((size_t)b * CIN + o0 + mb + r) * SEQ + l] = acc[j][r] + bias[r];
  }
}

// ---------------------------------------------------------------------------
extern "C" void kernel_launch(void* const* d_in, const int* in_sizes, int n_in,
                              void* d_out, int out_size, void* d_ws, size_t ws_size,
                              hipStream_t stream) {
  const float* x     = (const float*)d_in[0];   // [4,256,2048]
  const float* w_qkv = (const float*)d_in[1];   // [1536,256]
  const float* w_out = (const float*)d_in[2];   // [256,512]
  const float* b_out = (const float*)d_in[3];   // [256]
  float* y = (float*)d_out;                     // [4,256,2048]

  __bf16* p = (__bf16*)d_ws;
  __bf16* xT    = p;  p += (size_t)BATCH * SEQ * CIN;        //  4 MB
  __bf16* wq_bf = p;  p += (size_t)QKV_O * CIN;              //  0.75 MB
  __bf16* wo_bf = p;  p += (size_t)CIN * HIDDEN;             //  0.25 MB
  __bf16* Qw    = p;  p += (size_t)BATCH * HIDDEN * SEQ;     //  8 MB
  __bf16* Kw    = p;  p += (size_t)BATCH * HIDDEN * SEQ;     //  8 MB
  __bf16* Vt    = p;  p += (size_t)BATCH * HIDDEN * SEQ;     //  8 MB
  __bf16* attT  = p;                                         //  8 MB

  cvt_bf16_kernel<<<(QKV_O * CIN + 255) / 256, 256, 0, stream>>>(w_qkv, wq_bf,
                                                                 QKV_O * CIN);
  cvt_bf16_kernel<<<(CIN * HIDDEN + 255) / 256, 256, 0, stream>>>(w_out, wo_bf,
                                                                  CIN * HIDDEN);
  x_transpose_kernel<<<dim3(SEQ / 32, CIN / 32, BATCH), dim3(32, 8), 0, stream>>>(
      x, xT);
  qkv_proj_kernel<<<3072, 128, 0, stream>>>(xT, wq_bf, Qw, Kw, Vt);
  flash_attn_kernel<<<1024, 128, 0, stream>>>(Qw, Kw, Vt, attT);
  out_proj_kernel<<<512, 128, 0, stream>>>(attT, wo_bf, b_out, y);
}
